// MultiHeadAttention_24515673325887
// MI455X (gfx1250) — compile-verified, matrix operand order checked by reading, not run on hardware
//
#include <hip/hip_runtime.h>

#define DMODEL 1024
#define NHEADS 16
#define DK     64
#define BATCH  4
#define SEQ    2048
#define NTOK   (BATCH * SEQ)   // 8192

typedef __attribute__((ext_vector_type(16))) _Float16     v16h;
typedef __attribute__((ext_vector_type(8)))  float        v8f;
typedef __attribute__((ext_vector_type(4)))  unsigned int u32x4;
typedef __attribute__((ext_vector_type(8)))  int          i32x8;
typedef __attribute__((ext_vector_type(4)))  int          i32x4;

#define LOG2E 1.44269504088896340736f

// LDS slab geometry for TDM-staged A tiles: 16 rows x 64 halfs, padded to a
// 36-DWORD (72-half) row pitch so the 16 row-parallel lanes of an A-fragment
// read land in different LDS banks.
#define SLAB_ROW_H   72                 // halfs per padded row
#define SLAB_HALFS   (16 * SLAB_ROW_H)  // 1152 halfs = 2304 B per slab

__device__ __forceinline__ int laneId() { return (int)(threadIdx.x & 31); }

// WMMA wrapper: D = A(16x32 f16) * B(32x16 f16) + C(16x16 f32)
__device__ __forceinline__ v8f wmma16x16x32(v16h a, v16h b, v8f c) {
  return __builtin_amdgcn_wmma_f32_16x16x32_f16(false, a, false, b, (short)0, c,
                                                false, false);
}

// Load a 16x32 A-fragment from a row-major f16 matrix (leading dim ld, halfs).
// Also serves as the B-fragment loader when B = (row-major matrix)^T.
// ISA 7.12.2 (16-bit A 16x32): row = lane&15, g = lane>>4;
//   e<8 -> K = 8g+e ; e>=8 -> K = 16+8g+(e-8)
__device__ __forceinline__ v16h load_frag_rowmajor(const _Float16* base, int ld) {
  const int lane = laneId();
  const int r = lane & 15, g = lane >> 4;
  const _Float16* p = base + (size_t)r * ld + 8 * g;
  v16h f;
#pragma unroll
  for (int e = 0; e < 8; ++e) f[e] = p[e];
#pragma unroll
  for (int e = 0; e < 8; ++e) f[8 + e] = p[16 + e];
  return f;
}

// Load a 32x16 B-fragment from a K-major (row-major [K][N]) f16 matrix.
__device__ __forceinline__ v16h load_fragB_kmajor(const _Float16* base, int ld) {
  const int lane = laneId();
  const int c = lane & 15, g = lane >> 4;
  const _Float16* p = base + (size_t)(8 * g) * ld + c;
  v16h f;
#pragma unroll
  for (int e = 0; e < 8; ++e) f[e] = p[(size_t)e * ld];
  p += (size_t)16 * ld;
#pragma unroll
  for (int e = 0; e < 8; ++e) f[8 + e] = p[(size_t)e * ld];
  return f;
}

// ---- Tensor Data Mover: 2-D tile load (f16 elements) global -> LDS.
// Builds a D# per ISA ch.8: group0 {count=1, lds_addr, global_addr, type=2},
// group1 {data_size=2B, pad cfg, tensor dims, tile dims, dim0 stride}.
// Tile: tile_d0 halfs per row x tile_d1 rows; pad 4 DWORDs every 32 DWORDs
// (pad_interval code 4, pad_amount code 3) -> 36-DWORD LDS row pitch.
// amdgpu-toolchain (clang-23) builtin form: 6 args
//   (u32x4 g0, i32x8 g1, i32x4 g2, i32x4 g3, i32x8 extra, i32 cpol)
__device__ __forceinline__ void tdm_load_tile_f16(unsigned lds_addr,
                                                  const _Float16* gsrc,
                                                  unsigned tile_d0, unsigned tile_d1,
                                                  unsigned tensor_d0, unsigned tensor_d1,
                                                  unsigned d0_stride) {
  unsigned long long ga = (unsigned long long)(uintptr_t)gsrc;
  u32x4 g0;
  g0[0] = 1u;                                   // count=1 (valid user descriptor)
  g0[1] = lds_addr;                             // LDS byte address
  g0[2] = (unsigned)(ga & 0xffffffffu);         // global_addr[31:0]
  g0[3] = (unsigned)((ga >> 32) & 0x01ffffffu)  // global_addr[56:32]
          | (2u << 30);                         // type=2 ("image")
  i32x8 g1;
  g1[0] = (int)((1u << 16)        // data_size = 1 -> 2 bytes
                | (1u << 20)      // pad_enable
                | (4u << 22)      // pad_interval code 4 -> every 32 DWORDs
                | (3u << 25));    // pad_amount code 3 -> 4 DWORDs
  g1[1] = (int)((tensor_d0 & 0xffffu) << 16);                       // dim0[15:0]
  g1[2] = (int)(((tensor_d0 >> 16) & 0xffffu) |
                ((tensor_d1 & 0xffffu) << 16));                     // dim0 hi|dim1 lo
  g1[3] = (int)(((tensor_d1 >> 16) & 0xffffu) |
                ((tile_d0 & 0xffffu) << 16));                       // dim1 hi|tile0
  g1[4] = (int)(tile_d1 & 0xffffu);                                 // tile1, tile2=0
  g1[5] = (int)d0_stride;                                           // stride[31:0]
  g1[6] = 0;                                                        // stride[47:32]
  g1[7] = 0;
  i32x4 z4 = {0, 0, 0, 0};
  i32x8 z8 = {0, 0, 0, 0, 0, 0, 0, 0};
  __builtin_amdgcn_tensor_load_to_lds(g0, g1, z4, z4, z8, 0);
}

__device__ __forceinline__ unsigned lds_addr_of(const void* p) {
  return (unsigned)(uintptr_t)p;   // generic LDS addr: offset lives in [31:0]
}

// ---------------------------------------------------------------- converts
__global__ void cvt_f32_to_f16(const float* __restrict__ in,
                               _Float16* __restrict__ out, int n) {
  int i = blockIdx.x * blockDim.x + threadIdx.x;
  if (i < n) out[i] = (_Float16)in[i];
}

// ------------------------------------------------------------ GEMM core
// Computes one 16x64 tile of Amat(row-major MxK) @ W^T per wave, staging the
// A slab through LDS via TDM (double-buffered) and double-buffering B frags
// in registers so loads for step s+1 are in flight during the WMMAs of step s.
__device__ __forceinline__ void gemm_tile_16x64(const _Float16* __restrict__ Amat,
                                                const _Float16* __restrict__ W,
                                                int m0, int n0,
                                                _Float16* slab0, _Float16* slab1,
                                                v8f acc[4]) {
  const unsigned ls0 = lds_addr_of(slab0);
  const unsigned ls1 = lds_addr_of(slab1);
  const _Float16* arow = Amat + (size_t)m0 * DMODEL;

  // TDM prefetch of A chunk 0 (16 rows x 64 halfs)
  tdm_load_tile_f16(ls0, arow, 64, 16, DMODEL, NTOK, DMODEL);

  // B fragments for k-step 0
  v16h b0[4], b1[4];
#pragma unroll
  for (int j = 0; j < 4; ++j)
    b0[j] = load_frag_rowmajor(W + (size_t)(n0 + 16 * j) * DMODEL, DMODEL);

  for (int c = 0; c < DMODEL / 64; ++c) {
    const int kc = c * 64;
    _Float16* cur = (c & 1) ? slab1 : slab0;
    if (c + 1 < DMODEL / 64) {
      // async-prefetch next A chunk while this one computes
      tdm_load_tile_f16((c & 1) ? ls0 : ls1, arow + kc + 64, 64, 16, DMODEL,
                        NTOK, DMODEL);
      __builtin_prefetch((const void*)(W + (size_t)n0 * DMODEL + kc + 64), 0, 0);
      __builtin_amdgcn_s_wait_tensorcnt(1);   // chunk c landed; c+1 in flight
    } else {
      __builtin_amdgcn_s_wait_tensorcnt(0);
    }

    // ---- k-step A (kc): prefetch B for kc+32, consume b0
#pragma unroll
    for (int j = 0; j < 4; ++j)
      b1[j] = load_frag_rowmajor(W + (size_t)(n0 + 16 * j) * DMODEL + kc + 32,
                                 DMODEL);
    v16h a = load_frag_rowmajor(cur, SLAB_ROW_H);
#pragma unroll
    for (int j = 0; j < 4; ++j) acc[j] = wmma16x16x32(a, b0[j], acc[j]);

    // ---- k-step B (kc+32): prefetch B for kc+64, consume b1
    if (c + 1 < DMODEL / 64) {
#pragma unroll
      for (int j = 0; j < 4; ++j)
        b0[j] = load_frag_rowmajor(W + (size_t)(n0 + 16 * j) * DMODEL + kc + 64,
                                   DMODEL);
    }
    a = load_frag_rowmajor(cur + 32, SLAB_ROW_H);
#pragma unroll
    for (int j = 0; j < 4; ++j) acc[j] = wmma16x16x32(a, b1[j], acc[j]);
  }
}

// ---------------------------------------------------------------- QKV GEMM
__global__ void qkv_gemm(const _Float16* __restrict__ xh,
                         const _Float16* __restrict__ wqh,
                         const _Float16* __restrict__ wkh,
                         const _Float16* __restrict__ wvh,
                         _Float16* __restrict__ Qh,   // [B,H,N,DK] f16 (pre-scaled)
                         _Float16* __restrict__ Kh,   // [B,H,N,DK] f16
                         _Float16* __restrict__ Vh,   // [B,H,N,DK] f16
                         float* __restrict__ Kout,    // [B,H,N,DK] f32 (returned)
                         float* __restrict__ Vout) {  // [B,H,N,DK] f32 (returned)
  __shared__ _Float16 sA[8][2][SLAB_HALFS];
  const int proj = blockIdx.y;
  const _Float16* W = (proj == 0) ? wqh : (proj == 1) ? wkh : wvh;
  const int w = __builtin_amdgcn_readfirstlane((int)(threadIdx.x >> 5));
  const int wave = (int)blockIdx.x * 8 + w;
  const int nt = wave & 15;   // 16 n-tiles of 64
  const int mt = wave >> 4;   // 512 m-tiles of 16
  const int m0 = mt * 16, n0 = nt * 64;

  v8f acc[4] = {};
  gemm_tile_16x64(xh, W, m0, n0, &sA[w][0][0], &sA[w][1][0], acc);

  const int lane = laneId();
  const int col = lane & 15, g = lane >> 4;
#pragma unroll
  for (int j = 0; j < 4; ++j) {
#pragma unroll
    for (int v = 0; v < 8; ++v) {
      int m = m0 + v + 8 * g;        // token 0..8191
      int n = n0 + 16 * j + col;     // model dim 0..1023
      int b = m >> 11, t = m & (SEQ - 1);
      int hd = n >> 6, d = n & (DK - 1);
      size_t bhnd = ((size_t)(b * NHEADS + hd) * SEQ + t) * DK + d;
      float y = acc[j][v];
      if (proj == 0) {
        Qh[bhnd] = (_Float16)(y * 0.125f);   // fold 1/sqrt(64)
      } else if (proj == 1) {
        Kout[bhnd] = y; Kh[bhnd] = (_Float16)y;
      } else {
        Vout[bhnd] = y; Vh[bhnd] = (_Float16)y;
      }
    }
  }
}

// ---------------------------------------------------------------- attention
#define P_LD 40   // padded LDS row stride (halfs) to dodge bank conflicts
__global__ void flash_attn(const _Float16* __restrict__ Qh,
                           const _Float16* __restrict__ Kh,
                           const _Float16* __restrict__ Vh,
                           _Float16* __restrict__ valsh) {  // [B,N,DMODEL] f16
  __shared__ _Float16 sP[8][16 * P_LD];   // per-wave P staging (transpose)
  const int w = (int)(threadIdx.x >> 5);
  const int lane = laneId();
  const int col = lane & 15, g = lane >> 4;
  const int bh = blockIdx.x;              // b*16 + head
  const int b = bh >> 4, head = bh & 15;
  const int q0 = (blockIdx.y * 8 + w) * 16;
  const int qEnd = q0 + 15;

  const _Float16* Qb = Qh + ((size_t)bh * SEQ + q0) * DK;
  const _Float16* Kb = Kh + (size_t)bh * SEQ * DK;
  const _Float16* Vb = Vh + (size_t)bh * SEQ * DK;
  _Float16* myP = &sP[w][0];

  v16h aq0 = load_frag_rowmajor(Qb, DK);       // d 0..31
  v16h aq1 = load_frag_rowmajor(Qb + 32, DK);  // d 32..63

  float m[8], l[8];
  v8f o[4] = {};
#pragma unroll
  for (int v = 0; v < 8; ++v) { m[v] = -3.0e38f; l[v] = 0.0f; }

  for (int kBase = 0; kBase <= qEnd; kBase += 32) {
    // ---- S = Q K^T  (scale pre-folded into Qh), 16x32 block of scores
    v8f s0 = {}, s1 = {};
    {
      v16h b0 = load_frag_rowmajor(Kb + (size_t)kBase * DK, DK);
      v16h b1 = load_frag_rowmajor(Kb + (size_t)kBase * DK + 32, DK);
      v16h b2 = load_frag_rowmajor(Kb + (size_t)(kBase + 16) * DK, DK);
      v16h b3 = load_frag_rowmajor(Kb + (size_t)(kBase + 16) * DK + 32, DK);
      s0 = wmma16x16x32(aq0, b0, s0);
      s0 = wmma16x16x32(aq1, b1, s0);
      s1 = wmma16x16x32(aq0, b2, s1);
      s1 = wmma16x16x32(aq1, b3, s1);
    }
    // ---- causal mask (lane selects; EXEC stays all-1s)
#pragma unroll
    for (int v = 0; v < 8; ++v) {
      int rowg = q0 + v + 8 * g;
      s0[v] = (kBase + col > rowg) ? -3.0e30f : s0[v];
      s1[v] = (kBase + 16 + col > rowg) ? -3.0e30f : s1[v];
    }
    // ---- online softmax: row max across the 16 lanes holding one row
    float mb[8];
#pragma unroll
    for (int v = 0; v < 8; ++v) mb[v] = fmaxf(s0[v], s1[v]);
#pragma unroll
    for (int off = 1; off < 16; off <<= 1)
#pragma unroll
      for (int v = 0; v < 8; ++v)
        mb[v] = fmaxf(mb[v], __shfl_xor(mb[v], off, 16));

    float alpha[8], p0[8], p1[8], rs[8];
#pragma unroll
    for (int v = 0; v < 8; ++v) {
      float mn = fmaxf(m[v], mb[v]);
      alpha[v] = exp2f((m[v] - mn) * LOG2E);
      m[v] = mn;
      p0[v] = exp2f((s0[v] - mn) * LOG2E);
      p1[v] = exp2f((s1[v] - mn) * LOG2E);
      rs[v] = p0[v] + p1[v];
    }
#pragma unroll
    for (int off = 1; off < 16; off <<= 1)
#pragma unroll
      for (int v = 0; v < 8; ++v) rs[v] += __shfl_xor(rs[v], off, 16);
#pragma unroll
    for (int v = 0; v < 8; ++v) l[v] = l[v] * alpha[v] + rs[v];

    // ---- transpose P (C-layout -> A-layout) through this wave's LDS slice
#pragma unroll
    for (int v = 0; v < 8; ++v) {
      int row = v + 8 * g;
      myP[row * P_LD + col] = (_Float16)p0[v];
      myP[row * P_LD + 16 + col] = (_Float16)p1[v];
    }
    asm volatile("s_wait_dscnt 0" ::: "memory");  // cross-lane LDS visibility
    v16h aP = load_frag_rowmajor(myP, P_LD);      // P as 16x32 A-fragment

    // ---- O = O*alpha + P @ V
#pragma unroll
    for (int j = 0; j < 4; ++j) {
#pragma unroll
      for (int v = 0; v < 8; ++v) o[j][v] *= alpha[v];
      v16h bv = load_fragB_kmajor(Vb + (size_t)kBase * DK + 16 * j, DK);
      o[j] = wmma16x16x32(aP, bv, o[j]);
    }
  }

  // ---- normalize and store vals in [B, N, H*DK] layout for the out-proj
#pragma unroll
  for (int v = 0; v < 8; ++v) {
    float inv = 1.0f / l[v];
    int row = v + 8 * g;
    size_t base = ((size_t)b * SEQ + q0 + row) * DMODEL + head * DK;
#pragma unroll
    for (int j = 0; j < 4; ++j)
      valsh[base + 16 * j + col] = (_Float16)(o[j][v] * inv);
  }
}

// ---------------------------------------------------------------- out GEMM
__global__ void out_gemm(const _Float16* __restrict__ vh,
                         const _Float16* __restrict__ woh,
                         float* __restrict__ out) {
  __shared__ _Float16 sA[8][2][SLAB_HALFS];
  const int w = __builtin_amdgcn_readfirstlane((int)(threadIdx.x >> 5));
  const int wave = (int)blockIdx.x * 8 + w;
  const int nt = wave & 15, mt = wave >> 4;
  const int m0 = mt * 16, n0 = nt * 64;

  v8f acc[4] = {};
  gemm_tile_16x64(vh, woh, m0, n0, &sA[w][0][0], &sA[w][1][0], acc);

  const int lane = laneId();
  const int col = lane & 15, g = lane >> 4;
#pragma unroll
  for (int j = 0; j < 4; ++j)
#pragma unroll
    for (int v = 0; v < 8; ++v)
      out[(size_t)(m0 + v + 8 * g) * DMODEL + n0 + 16 * j + col] = acc[j][v];
}

// ---------------------------------------------------------------- launch
extern "C" void kernel_launch(void* const* d_in, const int* in_sizes, int n_in,
                              void* d_out, int out_size, void* d_ws, size_t ws_size,
                              hipStream_t stream) {
  const float* x  = (const float*)d_in[0];
  const float* wq = (const float*)d_in[1];
  const float* wk = (const float*)d_in[2];
  const float* wv = (const float*)d_in[3];
  const float* wo = (const float*)d_in[4];

  const size_t TOK_ELEMS = (size_t)NTOK * DMODEL;       // 8,388,608
  const size_t W_ELEMS   = (size_t)DMODEL * DMODEL;     // 1,048,576

  float* out  = (float*)d_out;                 // [B,N,DMODEL]
  float* Kout = out + TOK_ELEMS;               // [B,H,N,DK]
  float* Vout = Kout + TOK_ELEMS;              // [B,H,N,DK]

  _Float16* ws    = (_Float16*)d_ws;
  _Float16* xh    = ws;                        // TOK_ELEMS
  _Float16* wqh   = xh + TOK_ELEMS;            // W_ELEMS
  _Float16* wkh   = wqh + W_ELEMS;
  _Float16* wvh   = wkh + W_ELEMS;
  _Float16* woh   = wvh + W_ELEMS;
  _Float16* Qh    = woh + W_ELEMS;             // TOK_ELEMS (BHND)
  _Float16* Kh    = Qh + TOK_ELEMS;
  _Float16* Vh    = Kh + TOK_ELEMS;
  _Float16* valsh = Vh + TOK_ELEMS;            // TOK_ELEMS (B,N,D)

  const int CT = 256;
  cvt_f32_to_f16<<<(int)(TOK_ELEMS / CT), CT, 0, stream>>>(x, xh, (int)TOK_ELEMS);
  cvt_f32_to_f16<<<(int)(W_ELEMS / CT), CT, 0, stream>>>(wq, wqh, (int)W_ELEMS);
  cvt_f32_to_f16<<<(int)(W_ELEMS / CT), CT, 0, stream>>>(wk, wkh, (int)W_ELEMS);
  cvt_f32_to_f16<<<(int)(W_ELEMS / CT), CT, 0, stream>>>(wv, wvh, (int)W_ELEMS);
  cvt_f32_to_f16<<<(int)(W_ELEMS / CT), CT, 0, stream>>>(wo, woh, (int)W_ELEMS);

  // 8192 wave-tiles (512 m-tiles x 16 n-tiles), 8 waves/block -> 1024 blocks
  qkv_gemm<<<dim3(1024, 3), 256, 0, stream>>>(xh, wqh, wkh, wvh,
                                              Qh, Kh, Vh, Kout, Vout);
  // 64 (b,h) x 128 q-tiles; 8 q-tiles per block -> grid (64,16)
  flash_attn<<<dim3(BATCH * NHEADS, SEQ / (16 * 8)), 256, 0, stream>>>(Qh, Kh, Vh,
                                                                       valsh);
  out_gemm<<<1024, 256, 0, stream>>>(valsh, woh, out);
}
